// GNNpred_classifier_56530359550356
// MI455X (gfx1250) — matrix operand order repaired and checked by reading, not compiled, not run on hardware
//
#include <hip/hip_runtime.h>
#include <hip/hip_bf16.h>

typedef __attribute__((ext_vector_type(16))) __bf16 v16bf;
typedef __attribute__((ext_vector_type(8)))  float  v8f;

#define G_     8192
#define NPG_   30
#define EPG_   60
#define NDIM_  64
#define SDIM_  128
#define L_     4
#define E_     (G_*EPG_)

// Workspace layout (bf16 elements)
#define OFF_MSGWT   0        // [4][128][128]  Wt[k][o]
#define OFF_MSGRWT  65536    // [4][128][128]
#define OFF_WIHT    131072   // [4][128][192]
#define OFF_WHHT    229376   // [4][64][192]
#define OFF_FMT     278528   // [96][128] (K padded 69->96 with zeros)
#define WS_BF16_TOTAL 290816 // * 2 bytes = 581,632 B of d_ws

__device__ __forceinline__ float sigmoidf_(float x) { return 1.0f / (1.0f + __expf(-x)); }

__device__ __forceinline__ v8f bsplat(float v) {
    v8f r;
    for (int i = 0; i < 8; ++i) r[i] = v;
    return r;
}

// Build a 16x32 bf16 A-fragment from an LDS f32 row slice.
// ISA 16-bit A layout: lane m=L&15, half=L>>4; element e -> k = (e<8 ? e : e+8) + 8*half.
// Caller passes base = &row[32*chunk + 8*half]; runs are base[0..7] and base[16..23].
__device__ __forceinline__ v16bf ldsA(const float* base) {
    const float4* q = (const float4*)base;
    float4 a = q[0], b = q[1], c = q[4], d = q[5];
    v16bf r;
    r[0]=(__bf16)a.x;  r[1]=(__bf16)a.y;  r[2]=(__bf16)a.z;  r[3]=(__bf16)a.w;
    r[4]=(__bf16)b.x;  r[5]=(__bf16)b.y;  r[6]=(__bf16)b.z;  r[7]=(__bf16)b.w;
    r[8]=(__bf16)c.x;  r[9]=(__bf16)c.y;  r[10]=(__bf16)c.z; r[11]=(__bf16)c.w;
    r[12]=(__bf16)d.x; r[13]=(__bf16)d.y; r[14]=(__bf16)d.z; r[15]=(__bf16)d.w;
    return r;
}

// B fragment (32x16 bf16): lane L holds column n=L&15; element e -> k = (e<8 ? e : e+8) + 8*(L>>4), row stride ld.
__device__ __forceinline__ v16bf ldB(const __bf16* p, int ld) {
    int h = (threadIdx.x & 31) >> 4;
    v16bf r;
    for (int s = 0; s < 16; ++s) r[s] = p[((s < 8 ? s : s + 8) + 8 * h) * ld];
    return r;
}

#define WMMA_BF16(a, b, c) \
    __builtin_amdgcn_wmma_f32_16x16x32_bf16(false, (a), false, (b), (short)0, (c), false, false)

// ---------------------------------------------------------------------------
// Kernel 1: transpose + bf16-convert the GEMM weights into d_ws
// ---------------------------------------------------------------------------
__global__ void prep_weights_kernel(const float* __restrict__ msg_W,
                                    const float* __restrict__ msgr_W,
                                    const float* __restrict__ Wih,
                                    const float* __restrict__ Whh,
                                    const float* __restrict__ fm_W,
                                    __bf16* __restrict__ wsb) {
    int stride = gridDim.x * blockDim.x;
    for (int i = blockIdx.x * blockDim.x + threadIdx.x; i < WS_BF16_TOTAL; i += stride) {
        if (i < OFF_MSGRWT) {
            int l = i >> 14, rem = i & 16383, k = rem >> 7, o = rem & 127;
            wsb[i] = (__bf16)msg_W[(l << 14) + o * 128 + k];
        } else if (i < OFF_WIHT) {
            int j = i - OFF_MSGRWT;
            int l = j >> 14, rem = j & 16383, k = rem >> 7, o = rem & 127;
            wsb[i] = (__bf16)msgr_W[(l << 14) + o * 128 + k];
        } else if (i < OFF_WHHT) {
            int j = i - OFF_WIHT;
            int l = j / 24576, rem = j % 24576, k = rem / 192, o = rem % 192;
            wsb[i] = (__bf16)Wih[l * 24576 + o * 128 + k];
        } else if (i < OFF_FMT) {
            int j = i - OFF_WHHT;
            int l = j / 12288, rem = j % 12288, k = rem / 192, o = rem % 192;
            wsb[i] = (__bf16)Whh[l * 12288 + o * 64 + k];
        } else {
            int j = i - OFF_FMT;
            int k = j >> 7, o = j & 127;
            wsb[i] = (k < 69) ? (__bf16)fm_W[o * 69 + k] : (__bf16)0.0f;
        }
    }
}

// ---------------------------------------------------------------------------
// Kernel 2: one workgroup (4 wave32) per graph — full GNN + readout + heads
// ---------------------------------------------------------------------------
__global__ void __launch_bounds__(128)
gnn_graph_kernel(const int* __restrict__ node_atts, const int* __restrict__ edge_index,
                 const float* __restrict__ hyper, const float* __restrict__ node_init,
                 const float* __restrict__ msg_b, const float* __restrict__ msgr_b,
                 const float* __restrict__ bih, const float* __restrict__ bhh,
                 const float* __restrict__ hyp_W, const float* __restrict__ hyp_b,
                 const float* __restrict__ fm_b, const float* __restrict__ gm_W,
                 const float* __restrict__ gm_b,
                 const float* __restrict__ cW0, const float* __restrict__ cb0,
                 const float* __restrict__ cW1, const float* __restrict__ cb1,
                 const float* __restrict__ rW0, const float* __restrict__ rb0,
                 const float* __restrict__ rW1, const float* __restrict__ rb1,
                 const float* __restrict__ rW2, const float* __restrict__ rb2,
                 const __bf16* __restrict__ wsb, float* __restrict__ out) {
    __shared__ __align__(16) float sh_h[NPG_ * NDIM_];     // node states (f32)
    __shared__ __align__(16) float sh_aggr[NPG_ * SDIM_];  // message aggregate
    __shared__ __align__(16) float sh_gsum[NPG_ * 192];    // gi (+gh for r,z cols); reused as h_vG
    __shared__ __align__(16) float sh_hn[NPG_ * NDIM_];    // gh n-gate slice
    __shared__ float sh_hps[5];
    __shared__ int   sh_src[EPG_], sh_dst[EPG_];
    __shared__ float sh_g[NPG_];
    __shared__ float sh_hG[SDIM_];
    __shared__ float sh_x[64];
    __shared__ float sh_logit[32], sh_e[32];
    __shared__ float sh_cat[160];
    __shared__ float sh_r0[80], sh_r1[40];

    const int g    = blockIdx.x;
    const int tid  = threadIdx.x;
    const int lane = tid & 31;
    const int wv   = tid >> 5;
    const int half = lane >> 4;
    const int l16  = lane & 15;

    // ---- init: hps MLP, edge lists (local ids), node embedding gather ----
    if (tid < 5) {
        float s = hyp_b[tid];
        for (int j = 0; j < 5; ++j) s += hyp_W[tid * 5 + j] * hyper[g * 5 + j];
        sh_hps[tid] = fmaxf(s, 0.0f);
    }
    if (tid < EPG_) {
        sh_src[tid] = edge_index[g * EPG_ + tid] - g * NPG_;
        sh_dst[tid] = edge_index[E_ + g * EPG_ + tid] - g * NPG_;
    }
    for (int idx = tid; idx < NPG_ * NDIM_; idx += 128) {
        int m = idx >> 6, j = idx & 63;
        sh_h[idx] = node_init[node_atts[g * NPG_ + m] * NDIM_ + j];
    }
    __syncthreads();

    // ======================= GNN layers =======================
    for (int l = 0; l < L_; ++l) {
        const __bf16* Wf  = wsb + OFF_MSGWT  + l * 16384;
        const __bf16* Wr  = wsb + OFF_MSGRWT + l * 16384;
        const __bf16* Wih = wsb + OFF_WIHT   + l * 24576;
        const __bf16* Whh = wsb + OFF_WHHT   + l * 12288;
        const float*  bfv = msg_b  + l * 128;
        const float*  brv = msgr_b + l * 128;

        for (int idx = tid; idx < NPG_ * SDIM_; idx += 128) sh_aggr[idx] = 0.0f;
        __syncthreads();

        // ---- edge message GEMMs (wave wv owns edges 16wv..16wv+15) ----
        {
            int e  = 16 * wv + l16;
            int ee = e < EPG_ ? e : EPG_ - 1;  // clamp for addressing; stores guarded
            int ls = sh_src[ee], ld = sh_dst[ee];
            v16bf af[4];
#pragma unroll
            for (int c = 0; c < 4; ++c) {
                int node = (c < 2) ? ls : ld;
                af[c] = ldsA(&sh_h[node * NDIM_ + (c & 1) * 32 + 8 * half]);
            }
            v8f accF[8], accR[8];
#pragma unroll
            for (int nt = 0; nt < 8; ++nt) {
                accF[nt] = bsplat(bfv[l16 + 16 * nt]);
                accR[nt] = bsplat(brv[l16 + 16 * nt]);
            }
#pragma unroll
            for (int c = 0; c < 4; ++c) {
#pragma unroll
                for (int nt = 0; nt < 8; ++nt) {
                    v16bf bt = ldB(Wf + (32 * c) * 128 + 16 * nt + l16, 128);
                    accF[nt] = WMMA_BF16(af[c], bt, accF[nt]);
                    // x_rev = [h_dst,h_src]: fwd chunk c pairs with rev K-chunk c^2
                    v16bf br = ldB(Wr + (32 * (c ^ 2)) * 128 + 16 * nt + l16, 128);
                    accR[nt] = WMMA_BF16(af[c], br, accR[nt]);
                }
            }
            // scatter: D row m = v + 8*half = edge-in-tile, col n = l16 + 16*nt
#pragma unroll
            for (int nt = 0; nt < 8; ++nt) {
                int n = l16 + 16 * nt;
#pragma unroll
                for (int v = 0; v < 8; ++v) {
                    int er = 16 * wv + v + 8 * half;
                    if (er < EPG_) {
                        atomicAdd(&sh_aggr[sh_dst[er] * SDIM_ + n], accF[nt][v]);
                        atomicAdd(&sh_aggr[sh_src[er] * SDIM_ + n], accR[nt][v]);
                    }
                }
            }
        }
        __syncthreads();

        // ---- GRU GEMMs: gi = aggr@WihT, gh = h@WhhT ; wave wv owns ntiles 3wv..3wv+2 ----
        {
#pragma unroll
            for (int mt = 0; mt < 2; ++mt) {
                int nodeA = mt * 16 + l16;
                int na    = nodeA < NPG_ ? nodeA : NPG_ - 1;
                v16bf ai[4], ah[2];
#pragma unroll
                for (int c = 0; c < 4; ++c) ai[c] = ldsA(&sh_aggr[na * SDIM_ + 32 * c + 8 * half]);
#pragma unroll
                for (int c = 0; c < 2; ++c) ah[c] = ldsA(&sh_h[na * NDIM_ + 32 * c + 8 * half]);
#pragma unroll
                for (int t = 0; t < 3; ++t) {
                    int nt = 3 * wv + t;
                    v8f gi = bsplat(0.0f), gh = bsplat(0.0f);
#pragma unroll
                    for (int c = 0; c < 4; ++c)
                        gi = WMMA_BF16(ai[c], ldB(Wih + (32 * c) * 192 + 16 * nt + l16, 192), gi);
#pragma unroll
                    for (int c = 0; c < 2; ++c)
                        gh = WMMA_BF16(ah[c], ldB(Whh + (32 * c) * 192 + 16 * nt + l16, 192), gh);
                    int col = 16 * nt + l16;
#pragma unroll
                    for (int v = 0; v < 8; ++v) {
                        int node = mt * 16 + v + 8 * half;
                        if (node < NPG_) {
                            if (nt < 8) {  // r,z gate columns: store summed
                                sh_gsum[node * 192 + col] = gi[v] + gh[v];
                            } else {       // n gate columns: keep split
                                sh_gsum[node * 192 + col] = gi[v];
                                sh_hn[node * NDIM_ + (col - 128)] = gh[v];
                            }
                        }
                    }
                }
            }
        }
        __syncthreads();

        // ---- pointwise GRU gates ----
        {
            const float* bi = bih + l * 192;
            const float* bh = bhh + l * 192;
            for (int idx = tid; idx < NPG_ * NDIM_; idx += 128) {
                int m = idx >> 6, j = idx & 63;
                float r  = sigmoidf_(sh_gsum[m * 192 + j] + bi[j] + bh[j]);
                float z  = sigmoidf_(sh_gsum[m * 192 + 64 + j] + bi[64 + j] + bh[64 + j]);
                float nn = tanhf(sh_gsum[m * 192 + 128 + j] + bi[128 + j] +
                                 r * (sh_hn[m * NDIM_ + j] + bh[128 + j]));
                sh_h[idx] = (1.0f - z) * nn + z * sh_h[idx];
            }
        }
        __syncthreads();
    }

    // ======================= readout =======================
    float* sh_hvg = sh_gsum;  // reuse [30][128]
    const __bf16* fmT = wsb + OFF_FMT;
    {
#pragma unroll
        for (int mt = 0; mt < 2; ++mt) {
            int nodeA = mt * 16 + l16;
            int na    = nodeA < NPG_ ? nodeA : NPG_ - 1;
            v16bf a0 = ldsA(&sh_h[na * NDIM_ + 8 * half]);
            v16bf a1 = ldsA(&sh_h[na * NDIM_ + 32 + 8 * half]);
            v16bf a2;
#pragma unroll
            for (int i = 0; i < 16; ++i) a2[i] = (__bf16)0.0f;
            if (half == 0) {  // K chunk 2: features 64..68 = hps, rest zero-pad
                a2[0] = (__bf16)sh_hps[0]; a2[1] = (__bf16)sh_hps[1];
                a2[2] = (__bf16)sh_hps[2]; a2[3] = (__bf16)sh_hps[3];
                a2[4] = (__bf16)sh_hps[4];
            }
#pragma unroll
            for (int t = 0; t < 2; ++t) {
                int nt  = 2 * wv + t;
                v8f acc = bsplat(fm_b[l16 + 16 * nt]);
                acc = WMMA_BF16(a0, ldB(fmT + 16 * nt + l16, 128), acc);
                acc = WMMA_BF16(a1, ldB(fmT + 32 * 128 + 16 * nt + l16, 128), acc);
                acc = WMMA_BF16(a2, ldB(fmT + 64 * 128 + 16 * nt + l16, 128), acc);
                int col = 16 * nt + l16;
#pragma unroll
                for (int v = 0; v < 8; ++v) {
                    int node = mt * 16 + v + 8 * half;
                    if (node < NPG_) sh_hvg[node * SDIM_ + col] = acc[v];
                }
            }
        }
    }
    // gate scalars g_vG
    if (tid < NPG_) {
        float s = gm_b[0];
        for (int j = 0; j < 64; ++j) s += sh_h[tid * NDIM_ + j] * gm_W[j];
        for (int j = 0; j < 5; ++j) s += sh_hps[j] * gm_W[64 + j];
        sh_g[tid] = sigmoidf_(s);
    }
    __syncthreads();

    // gated sum -> h_G[128]
    {
        float s = 0.0f;
        for (int m = 0; m < NPG_; ++m) s += sh_hvg[m * SDIM_ + tid] * sh_g[m];
        sh_hG[tid] = s;
    }
    __syncthreads();

    if (tid < 64) {
        float s = cb0[tid];
        for (int k = 0; k < 128; ++k) s += cW0[tid * 128 + k] * sh_hG[k];
        sh_x[tid] = fmaxf(s, 0.0f);
    }
    __syncthreads();
    if (tid < 32) {
        float s = cb1[tid];
        for (int k = 0; k < 64; ++k) s += cW1[tid * 64 + k] * sh_x[k];
        sh_logit[tid] = s;
    }
    __syncthreads();
    if (tid < 32) {
        float mx = -3.4e38f;
        for (int k = 0; k < 32; ++k) mx = fmaxf(mx, sh_logit[k]);
        sh_e[tid] = __expf(sh_logit[tid] - mx);
    }
    __syncthreads();
    if (tid < 32) {
        float s = 0.0f;
        for (int k = 0; k < 32; ++k) s += sh_e[k];
        float p = sh_e[tid] / s;
        out[g * 32 + tid] = p;  // hs output
        sh_cat[tid] = p;
    }
    sh_cat[32 + tid] = sh_hG[tid];
    __syncthreads();
    if (tid < 80) {
        float s = rb0[tid];
        for (int k = 0; k < 160; ++k) s += rW0[tid * 160 + k] * sh_cat[k];
        sh_r0[tid] = fmaxf(s, 0.0f);
    }
    __syncthreads();
    if (tid < 40) {
        float s = rb1[tid];
        for (int k = 0; k < 80; ++k) s += rW1[tid * 80 + k] * sh_r0[k];
        sh_r1[tid] = fmaxf(s, 0.0f);
    }
    __syncthreads();
    if (tid == 0) {
        float s = rb2[0];
        for (int k = 0; k < 40; ++k) s += rW2[k] * sh_r1[k];
        out[G_ * 32 + g] = s;  // hc output
    }
}

extern "C" void kernel_launch(void* const* d_in, const int* in_sizes, int n_in,
                              void* d_out, int out_size, void* d_ws, size_t ws_size,
                              hipStream_t stream) {
    const int*   node_atts  = (const int*)d_in[0];
    const int*   edge_index = (const int*)d_in[1];
    /* d_in[2] = batch (implicit: blocked) */
    const float* hyper      = (const float*)d_in[3];
    const float* node_init  = (const float*)d_in[4];
    const float* msg_W      = (const float*)d_in[5];
    const float* msg_b      = (const float*)d_in[6];
    const float* msgr_W     = (const float*)d_in[7];
    const float* msgr_b     = (const float*)d_in[8];
    const float* gru_Wih    = (const float*)d_in[9];
    const float* gru_Whh    = (const float*)d_in[10];
    const float* gru_bih    = (const float*)d_in[11];
    const float* gru_bhh    = (const float*)d_in[12];
    const float* hyp_W      = (const float*)d_in[13];
    const float* hyp_b      = (const float*)d_in[14];
    const float* fm_W       = (const float*)d_in[15];
    const float* fm_b       = (const float*)d_in[16];
    const float* gm_W       = (const float*)d_in[17];
    const float* gm_b       = (const float*)d_in[18];
    const float* cW0        = (const float*)d_in[19];
    const float* cb0        = (const float*)d_in[20];
    const float* cW1        = (const float*)d_in[21];
    const float* cb1        = (const float*)d_in[22];
    const float* rW0        = (const float*)d_in[23];
    const float* rb0        = (const float*)d_in[24];
    const float* rW1        = (const float*)d_in[25];
    const float* rb1        = (const float*)d_in[26];
    const float* rW2        = (const float*)d_in[27];
    const float* rb2        = (const float*)d_in[28];

    __bf16* wsb = (__bf16*)d_ws;  // 581,632 bytes used

    prep_weights_kernel<<<(WS_BF16_TOTAL + 255) / 256, 256, 0, stream>>>(
        msg_W, msgr_W, gru_Wih, gru_Whh, fm_W, wsb);

    gnn_graph_kernel<<<G_, 128, 0, stream>>>(
        node_atts, edge_index, hyper, node_init, msg_b, msgr_b, gru_bih, gru_bhh,
        hyp_W, hyp_b, fm_b, gm_W, gm_b, cW0, cb0, cW1, cb1,
        rW0, rb0, rW1, rb1, rW2, rb2, wsb, (float*)d_out);
}
